// AttentionLanguageModel_37065567764861
// MI455X (gfx1250) — compile-verified, matrix operand order checked by reading, not run on hardware
//
#include <hip/hip_runtime.h>

// ---------------------------------------------------------------------------
// GPT-style forward pass for gfx1250 (MI455X).
//  - residual stream h kept in f32
//  - all GEMMs / attention matmuls in bf16 via v_wmma_f32_16x16x32_bf16
//  - global->LDS staging of bf16 tiles via GLOBAL_LOAD_ASYNC_TO_LDS_B128
//    (ASYNCcnt) when the builtin is available; sync fallback otherwise
//  - head GEMV kept f32 (bandwidth-bound, 2 live rows), b128 loads per lane
// ---------------------------------------------------------------------------

typedef __bf16 bf16;
typedef __attribute__((ext_vector_type(16))) __bf16 v16bf;
typedef __attribute__((ext_vector_type(8)))  float  v8f;
typedef __attribute__((ext_vector_type(4)))  int    v4i;
typedef __attribute__((ext_vector_type(4)))  unsigned int v4u;

#define WMMA_BF16(a, b, c) \
  __builtin_amdgcn_wmma_f32_16x16x32_bf16(false, (a), false, (b), (short)0, (c), false, false)

#if __has_builtin(__builtin_amdgcn_global_load_async_to_lds_b128) && \
    __has_builtin(__builtin_amdgcn_s_wait_asynccnt)
#define ASYNC_CP 1
#else
#define ASYNC_CP 0
#endif

// 16-byte global->LDS copy at compile-time byte offset OFF.
template <int OFF>
__device__ __forceinline__ void cp_async16(bf16* lds, const bf16* glb) {
#if ASYNC_CP
  bf16* g = const_cast<bf16*>(glb);   // builtin params are non-const typed ptrs
  __builtin_amdgcn_global_load_async_to_lds_b128(
      (__attribute__((address_space(1))) v4i*)g,
      (__attribute__((address_space(3))) v4i*)lds, OFF, 0);
#else
  *(v4u*)((char*)lds + OFF) = *(const v4u*)((const char*)glb + OFF);
#endif
}

__device__ __forceinline__ void wait_async() {
#if ASYNC_CP
  __builtin_amdgcn_s_wait_asynccnt(0);
#endif
}

static constexpr int D_   = 1024;   // embed dim
static constexpr int T_   = 1024;   // context
static constexpr int B_   = 2;      // batch
static constexpr int H_   = 16;     // heads
static constexpr int HD_  = 64;     // head dim
static constexpr int V_   = 32000;  // vocab
static constexpr int QLD  = 3072;   // qkv row stride (3*D)
static constexpr int L_   = 6;

// ---------------------------------------------------------------------------
// 1) token + positional embedding -> h (f32)
// ---------------------------------------------------------------------------
__global__ __launch_bounds__(256)
void embed_kernel(const int* __restrict__ tokens, const float* __restrict__ te,
                  const float* __restrict__ pe, float* __restrict__ h)
{
  int bt  = blockIdx.x;            // b*T + t
  int t   = bt & (T_ - 1);
  int tok = tokens[bt];
  const float* tr = te + (size_t)tok * D_;
  const float* pr = pe + (size_t)t   * D_;
  float*       hr = h  + (size_t)bt  * D_;
  for (int d = threadIdx.x; d < D_; d += 256) hr[d] = tr[d] + pr[d];
}

// ---------------------------------------------------------------------------
// 2) LayerNorm row kernel, f32 in -> bf16 out (one block per row)
// ---------------------------------------------------------------------------
__global__ __launch_bounds__(256)
void ln_kernel(const float* __restrict__ x, const float* __restrict__ w,
               const float* __restrict__ bb, bf16* __restrict__ out)
{
  __shared__ float rs_[8], rq_[8];
  int row = blockIdx.x;
  const float* xr = x + (size_t)row * D_;
  float s = 0.f, q = 0.f;
  for (int d = threadIdx.x; d < D_; d += 256) { float v = xr[d]; s += v; q += v * v; }
#pragma unroll
  for (int o = 16; o > 0; o >>= 1) { s += __shfl_xor(s, o, 32); q += __shfl_xor(q, o, 32); }
  if ((threadIdx.x & 31) == 0) { rs_[threadIdx.x >> 5] = s; rq_[threadIdx.x >> 5] = q; }
  __syncthreads();
  s = 0.f; q = 0.f;
#pragma unroll
  for (int i = 0; i < 8; i++) { s += rs_[i]; q += rq_[i]; }
  float mu  = s * (1.f / D_);
  float var = q * (1.f / D_) - mu * mu;
  float inv = rsqrtf(var + 1e-5f);
  bf16* orow = out + (size_t)row * D_;
  for (int d = threadIdx.x; d < D_; d += 256)
    orow[d] = (bf16)((xr[d] - mu) * inv * w[d] + bb[d]);
}

// Final LN: rows t = T-1 only, f32 out. blockIdx.x = b.
__global__ __launch_bounds__(256)
void lnf_kernel(const float* __restrict__ h, const float* __restrict__ w,
                const float* __restrict__ bb, float* __restrict__ out)
{
  __shared__ float rs_[8], rq_[8];
  int b = blockIdx.x;
  const float* xr = h + ((size_t)b * T_ + (T_ - 1)) * D_;
  float s = 0.f, q = 0.f;
  for (int d = threadIdx.x; d < D_; d += 256) { float v = xr[d]; s += v; q += v * v; }
#pragma unroll
  for (int o = 16; o > 0; o >>= 1) { s += __shfl_xor(s, o, 32); q += __shfl_xor(q, o, 32); }
  if ((threadIdx.x & 31) == 0) { rs_[threadIdx.x >> 5] = s; rq_[threadIdx.x >> 5] = q; }
  __syncthreads();
  s = 0.f; q = 0.f;
#pragma unroll
  for (int i = 0; i < 8; i++) { s += rs_[i]; q += rq_[i]; }
  float mu  = s * (1.f / D_);
  float var = q * (1.f / D_) - mu * mu;
  float inv = rsqrtf(var + 1e-5f);
  float* orow = out + (size_t)b * D_;
  for (int d = threadIdx.x; d < D_; d += 256)
    orow[d] = (xr[d] - mu) * inv * w[d] + bb[d];
}

// ---------------------------------------------------------------------------
// 3) WMMA GEMM: C(M=2048, N) = A(bf16, M x 1024) @ W(f32->bf16, 1024 x N) + bias
//    Block = 256 threads (8 waves), tile 128x128, BK=32.
//    Wave w owns rows w*16..+15 x all 128 cols -> 8 WMMAs per k-step,
//    A fragment reused 8x. A tile staged via async b128 copies.
//    mode 0: store bf16 to outb (qkv buffer, ldc = N)
//    mode 1: residual epilogue: hbuf += C (f32), ldc = N
// ---------------------------------------------------------------------------
__global__ __launch_bounds__(256)
void gemm_kernel(const bf16* __restrict__ A, const float* __restrict__ W,
                 const float* __restrict__ bias, int N,
                 bf16* __restrict__ outb, float* __restrict__ hbuf, int mode)
{
  const int K = 1024;
  __shared__ __align__(32) bf16 Asub[128][32];   // [m][k]   8 KB
  __shared__ __align__(32) bf16 BsubT[128][32];  // [n][k]   8 KB (transposed)

  int tid  = threadIdx.x;
  int wave = tid >> 5, lane = tid & 31;
  int half = lane >> 4, l16 = lane & 15;
  int m0 = blockIdx.x * 128;
  int n0 = blockIdx.y * 128;

  v8f acc[8] = {};

  for (int k0 = 0; k0 < K; k0 += 32) {
    __syncthreads();
    {   // stage A tile: 128x32 bf16, 32 bytes per thread via async b128 x2
      int r = tid >> 1, hh = tid & 1;
      const bf16* ga = A + (size_t)(m0 + r) * K + k0 + hh * 16;
      bf16*       la = &Asub[r][hh * 16];
      cp_async16<0>(la, ga);
      cp_async16<16>(la, ga);
    }
    {   // stage W tile: 32x128 f32 -> bf16, transposed into [n][k]
      int r = tid >> 3, qq = tid & 7;
      const float* wp = W + (size_t)(k0 + r) * N + n0 + qq * 16;
      if (k0 + 32 < K)  // prefetch next weight tile (global_prefetch_b8)
        __builtin_prefetch(wp + (size_t)32 * N, 0, 1);
#pragma unroll
      for (int j = 0; j < 16; j++) BsubT[qq * 16 + j][r] = (bf16)wp[j];
    }
    wait_async();
    __syncthreads();

    // A fragment: lane -> row (wave*16 + l16), elems k = half*16 + e
    v16bf af = *(const v16bf*)&Asub[wave * 16 + l16][half * 16];
#pragma unroll
    for (int c = 0; c < 8; c++) {
      v16bf bfg = *(const v16bf*)&BsubT[c * 16 + l16][half * 16];
      acc[c] = WMMA_BF16(af, bfg, acc[c]);
    }
  }

  // epilogue: C layout row = e + 8*half, col = l16 (per 16x16 tile)
#pragma unroll
  for (int c = 0; c < 8; c++) {
    int n = n0 + c * 16 + l16;
    float bv = bias[n];
#pragma unroll
    for (int e = 0; e < 8; e++) {
      int m = m0 + wave * 16 + e + 8 * half;
      float v = acc[c][e] + bv;
      if (mode == 0) {
        outb[(size_t)m * N + n] = (bf16)v;
      } else {
        hbuf[(size_t)m * N + n] += v;   // residual add into h
      }
    }
  }
}

// ---------------------------------------------------------------------------
// 4) Flash attention with WMMA. Block = 128 threads (4 waves), one block per
//    (q-tile of 64 rows, head, batch). Wave w owns q rows qt*64 + w*16 .. +15.
//    kb loop over key tiles of 32 (causal bound), online softmax.
//    K tile staged via async b128; V tile transposed through VGPRs.
// ---------------------------------------------------------------------------
__global__ __launch_bounds__(128)
void attn_kernel(const bf16* __restrict__ qkv, bf16* __restrict__ obuf)
{
  __shared__ __align__(32) bf16 Ks[32][64];      // key tile   [key][hd]
  __shared__ __align__(32) bf16 Vt[64][32];      // value tile [hd][key] (transposed)
  __shared__ __align__(32) bf16 Ps[4][16][32];   // per-wave P scratch [q][key]

  int tid  = threadIdx.x;
  int wave = tid >> 5, lane = tid & 31;
  int half = lane >> 4, l16 = lane & 15;
  int qt = blockIdx.x, hh = blockIdx.y, b = blockIdx.z;

  const bf16* Qb = qkv + hh * HD_;
  const bf16* Kb = qkv + D_ + hh * HD_;
  const bf16* Vb = qkv + 2 * D_ + hh * HD_;

  // Q fragments: lane -> q row (l16), elems hd = step*32 + half*16 + e
  int qrow = qt * 64 + wave * 16 + l16;
  size_t qoff = (size_t)(b * T_ + qrow) * QLD;
  v16bf aq0 = *(const v16bf*)(Qb + qoff + half * 16);
  v16bf aq1 = *(const v16bf*)(Qb + qoff + 32 + half * 16);

  v8f oc[4] = {};
  float rmax[8], rsum[8];
#pragma unroll
  for (int e = 0; e < 8; e++) { rmax[e] = -1e30f; rsum[e] = 0.f; }

  const float sc = 0.125f;   // 1/sqrt(64)
  int nkb = qt * 2 + 2;      // causal: keys up to qt*64+63

  for (int kb = 0; kb < nkb; kb++) {
    __syncthreads();
    {   // stage K (async, as-is) and V (transposed) tiles: 32 keys x 64 dims
      int r = tid >> 2, qq = tid & 3;
      size_t roff = (size_t)(b * T_ + kb * 32 + r) * QLD;
      const bf16* gk = Kb + roff + qq * 16;
      bf16*       lk = &Ks[r][qq * 16];
      cp_async16<0>(lk, gk);
      cp_async16<16>(lk, gk);
      v16bf vv = *(const v16bf*)(Vb + roff + qq * 16);
#pragma unroll
      for (int j = 0; j < 16; j++) Vt[qq * 16 + j][r] = vv[j];
    }
    wait_async();
    __syncthreads();

    // S = Q (16x64) @ K^T : two 16-key subtiles, 2 hd-steps each
    v8f s0 = {}, s1 = {};
    {
      v16bf bk;
      bk = *(const v16bf*)&Ks[l16][half * 16];           s0 = WMMA_BF16(aq0, bk, s0);
      bk = *(const v16bf*)&Ks[l16][32 + half * 16];      s0 = WMMA_BF16(aq1, bk, s0);
      bk = *(const v16bf*)&Ks[16 + l16][half * 16];      s1 = WMMA_BF16(aq0, bk, s1);
      bk = *(const v16bf*)&Ks[16 + l16][32 + half * 16]; s1 = WMMA_BF16(aq1, bk, s1);
    }

    // online softmax; row stats reduced across the 16 lanes of each half
    float p0[8], p1[8], alpha[8];
#pragma unroll
    for (int e = 0; e < 8; e++) {
      int m   = qt * 64 + wave * 16 + e + 8 * half;   // global q row
      int k0g = kb * 32 + l16;
      int k1g = kb * 32 + 16 + l16;
      float v0 = (k0g <= m) ? s0[e] * sc : -1e30f;
      float v1 = (k1g <= m) ? s1[e] * sc : -1e30f;
      float mx = fmaxf(v0, v1);
#pragma unroll
      for (int o = 1; o < 16; o <<= 1) mx = fmaxf(mx, __shfl_xor(mx, o, 32));
      float mn = fmaxf(rmax[e], mx);
      float e0 = __expf(v0 - mn), e1 = __expf(v1 - mn);
      float ls = e0 + e1;
#pragma unroll
      for (int o = 1; o < 16; o <<= 1) ls += __shfl_xor(ls, o, 32);
      alpha[e] = __expf(rmax[e] - mn);
      rsum[e]  = rsum[e] * alpha[e] + ls;
      rmax[e]  = mn;
      p0[e] = e0; p1[e] = e1;
    }
    // rescale O accumulators
#pragma unroll
    for (int nt = 0; nt < 4; nt++)
#pragma unroll
      for (int e = 0; e < 8; e++) oc[nt][e] *= alpha[e];

    // P (C-layout) -> LDS -> A-layout fragment (per-wave scratch, no barrier)
#pragma unroll
    for (int e = 0; e < 8; e++) {
      Ps[wave][e + 8 * half][l16]      = (bf16)p0[e];
      Ps[wave][e + 8 * half][16 + l16] = (bf16)p1[e];
    }
    v16bf pa = *(const v16bf*)&Ps[wave][l16][half * 16];

    // O += P (16x32) @ V (32x64): B frag from transposed Vt, contiguous
#pragma unroll
    for (int nt = 0; nt < 4; nt++) {
      v16bf bv = *(const v16bf*)&Vt[nt * 16 + l16][half * 16];
      oc[nt] = WMMA_BF16(pa, bv, oc[nt]);
    }
  }

  // epilogue: O /= rowsum, store bf16 to o buffer (B*T, D) at col h*64+hd
#pragma unroll
  for (int e = 0; e < 8; e++) {
    float inv = 1.f / rsum[e];
    int t = qt * 64 + wave * 16 + e + 8 * half;
    bf16* orow = obuf + (size_t)(b * T_ + t) * D_ + hh * HD_;
#pragma unroll
    for (int nt = 0; nt < 4; nt++)
      orow[nt * 16 + l16] = (bf16)(oc[nt][e] * inv);
  }
}

// ---------------------------------------------------------------------------
// 5) Head GEMV: logits(b, v) = xf(b,:) . head_w(:, v) + head_b(v)
//    Bandwidth-bound on 131 MB of head_w: one float4 (b128) per lane per k.
//    Block 64 threads; each thread owns 4 consecutive vocab columns.
// ---------------------------------------------------------------------------
__global__ __launch_bounds__(64)
void head_kernel(const float* __restrict__ xf, const float* __restrict__ hw,
                 const float* __restrict__ hb, float* __restrict__ out)
{
  __shared__ float xs[D_];
  int b = blockIdx.y;
  for (int d = threadIdx.x; d < D_; d += 64) xs[d] = xf[(size_t)b * D_ + d];
  __syncthreads();
  int v4 = (blockIdx.x * 64 + threadIdx.x) * 4;
  float4 acc = *(const float4*)&hb[v4];
  for (int d = 0; d < D_; d++) {
    float x = xs[d];
    float4 w = *(const float4*)&hw[(size_t)d * V_ + v4];
    acc.x += x * w.x; acc.y += x * w.y; acc.z += x * w.z; acc.w += x * w.w;
  }
  *(float4*)&out[(size_t)b * V_ + v4] = acc;
}

// ---------------------------------------------------------------------------
// launch
// ---------------------------------------------------------------------------
extern "C" void kernel_launch(void* const* d_in, const int* in_sizes, int n_in,
                              void* d_out, int out_size, void* d_ws, size_t ws_size,
                              hipStream_t stream)
{
  const int*   tokens    = (const int*)  d_in[0];
  const float* tok_embed = (const float*)d_in[1];
  const float* pos_embed = (const float*)d_in[2];
  const float* qkv_w     = (const float*)d_in[3];
  const float* qkv_b     = (const float*)d_in[4];
  const float* out_w     = (const float*)d_in[5];
  const float* out_b     = (const float*)d_in[6];
  const float* ln_w      = (const float*)d_in[7];
  const float* ln_b      = (const float*)d_in[8];
  const float* lnf_w     = (const float*)d_in[9];
  const float* lnf_b     = (const float*)d_in[10];
  const float* head_w    = (const float*)d_in[11];
  const float* head_b    = (const float*)d_in[12];

  // workspace layout (~28 MB total)
  char* ws = (char*)d_ws;
  float* h    = (float*)(ws);                        //  8 MB  (B,T,D) f32 residual
  bf16*  xln  = (bf16*) (ws + (size_t)(8  << 20));   //  4 MB  (B*T,D) bf16
  bf16*  qkvb = (bf16*) (ws + (size_t)(12 << 20));   // 12 MB  (B*T,3D) bf16
  bf16*  ob   = (bf16*) (ws + (size_t)(24 << 20));   //  4 MB  (B*T,D) bf16
  float* xf   = (float*)(ws + (size_t)(28 << 20));   //  8 KB  (B,D) f32

  embed_kernel<<<B_ * T_, 256, 0, stream>>>(tokens, tok_embed, pos_embed, h);

  for (int l = 0; l < L_; l++) {
    ln_kernel<<<B_ * T_, 256, 0, stream>>>(h, ln_w + (size_t)l * D_, ln_b + (size_t)l * D_, xln);

    dim3 g1(B_ * T_ / 128, QLD / 128);   // (16, 24)
    gemm_kernel<<<g1, 256, 0, stream>>>(xln, qkv_w + (size_t)l * D_ * QLD,
                                        qkv_b + (size_t)l * QLD, QLD, qkvb, nullptr, 0);

    dim3 ga(T_ / 64, H_, B_);            // (16, 16, 2)
    attn_kernel<<<ga, 128, 0, stream>>>(qkvb, ob);

    dim3 g2(B_ * T_ / 128, D_ / 128);    // (16, 8)
    gemm_kernel<<<g2, 256, 0, stream>>>(ob, out_w + (size_t)l * D_ * D_,
                                        out_b + (size_t)l * D_, D_, nullptr, h, 1);
  }

  lnf_kernel<<<B_, 256, 0, stream>>>(h, lnf_w, lnf_b, xf);
  head_kernel<<<dim3(V_ / 256, B_), 64, 0, stream>>>(xf, head_w, head_b, (float*)d_out);
}
